// ManualRGCNConv_11003706212538
// MI455X (gfx1250) — matrix operand order repaired and checked by reading, not compiled, not run on hardware
//
#include <hip/hip_runtime.h>
#include <stdint.h>

#define NN 20000      // nodes
#define NE 320000     // edges
#define NR 8          // relations
#define DD 512        // feature dim (in == out)

typedef __attribute__((ext_vector_type(16))) __bf16 v16bf;
typedef __attribute__((ext_vector_type(8)))  float  v8f;

struct U4x2 { uint4 a, b; };

__device__ __forceinline__ v16bf mkfrag(uint4 a, uint4 b) {
  U4x2 t; t.a = a; t.b = b;
  return __builtin_bit_cast(v16bf, t);
}

// round-to-nearest-even f32 -> bf16 (finite inputs)
__device__ __forceinline__ unsigned short f2bf_rne(float f) {
  unsigned int u = __builtin_bit_cast(unsigned int, f);
  u += 0x7fffu + ((u >> 16) & 1u);
  return (unsigned short)(u >> 16);
}
__device__ __forceinline__ float bf2f(unsigned short h) {
  unsigned int u = ((unsigned int)h) << 16;
  return __builtin_bit_cast(float, u);
}

// ---------------------------------------------------------------------------
// Convert W_self (block 0) and W_rel[r] (blocks 1..8) to hi/lo bf16 pairs.
// Layout preserved: [block][out][in], in-dim contiguous.
// ---------------------------------------------------------------------------
__global__ void convw_kernel(const float* __restrict__ wself,
                             const float* __restrict__ wrel,
                             unsigned short* __restrict__ Wh,
                             unsigned short* __restrict__ Wl) {
  size_t i = (size_t)blockIdx.x * blockDim.x + threadIdx.x;
  const size_t tot = (size_t)(NR + 1) * DD * DD;
  if (i >= tot) return;
  float w = (i < (size_t)DD * DD) ? wself[i] : wrel[i - (size_t)DD * DD];
  unsigned short h = f2bf_rne(w);
  unsigned short l = f2bf_rne(w - bf2f(h));
  Wh[i] = h;
  Wl[i] = l;
}

// ---------------------------------------------------------------------------
// Edge degree count per (relation, dst)
// ---------------------------------------------------------------------------
__global__ void count_kernel(const int* __restrict__ ei,
                             const int* __restrict__ et,
                             int* __restrict__ cnt) {
  int e = blockIdx.x * blockDim.x + threadIdx.x;
  if (e >= NE) return;
  int r = et[e];
  int dst = ei[NE + e];
  atomicAdd(&cnt[r * NN + dst], 1);
}

__global__ void inv_kernel(const int* __restrict__ cnt,
                           float* __restrict__ inv) {
  int i = blockIdx.x * blockDim.x + threadIdx.x;
  if (i >= NR * NN) return;
  int c = cnt[i];
  inv[i] = 1.0f / (float)(c > 0 ? c : 1);
}

// ---------------------------------------------------------------------------
// Scatter: one wave (32 lanes) per edge. Edges of relation `rel` add
// x[src] * inv_cnt[rel,dst] into agg[dst].
// ---------------------------------------------------------------------------
__global__ __launch_bounds__(256) void scatter_kernel(
    const float* __restrict__ x, const int* __restrict__ ei,
    const int* __restrict__ et, const float* __restrict__ inv,
    float* __restrict__ agg, int rel) {
  int wave = (blockIdx.x * blockDim.x + threadIdx.x) >> 5;
  int lane = threadIdx.x & 31;
  if (wave >= NE) return;
  if (et[wave] != rel) return;
  int src = ei[wave];
  int dst = ei[NE + wave];
  float s = inv[rel * NN + dst];
  const float4* xs = (const float4*)(x + (size_t)src * DD);
  float* ag = agg + (size_t)dst * DD;
#pragma unroll
  for (int c = 0; c < 4; ++c) {
    float4 v = xs[lane + c * 32];
    int j = (lane + c * 32) * 4;
    atomicAdd(ag + j + 0, v.x * s);
    atomicAdd(ag + j + 1, v.y * s);
    atomicAdd(ag + j + 2, v.z * s);
    atomicAdd(ag + j + 3, v.w * s);
  }
}

// ---------------------------------------------------------------------------
// GEMM: out[32 x 512 tile] (+)= A[32 x 512] @ W^T, W is [512 out][512 in],
// bf16-split (Ah*Bh + Ah*Bl + Al*Bh) with f32 WMMA accumulation.
// Block: 256 threads = 8 waves; wave w covers columns [w*64, w*64+64).
// ---------------------------------------------------------------------------
__global__ __launch_bounds__(256) void rgcn_gemm(
    const float* __restrict__ A, const unsigned short* __restrict__ Wh,
    const unsigned short* __restrict__ Wl, float* __restrict__ out,
    const float* __restrict__ bias, int accumulate) {
  __shared__ unsigned short Ah[32][40];  // +8 pad: conflict-free b128 reads
  __shared__ unsigned short Al[32][40];

  const int tid  = threadIdx.x;
  const int wv   = tid >> 5;        // wave 0..7 -> N strip
  const int lane = tid & 31;
  const int l16  = lane & 15;
  const int hi   = lane >> 4;       // half-wave select
  const long mBase = (long)blockIdx.x * 32;

  v8f acc[2][4];
#pragma unroll
  for (int s = 0; s < 2; ++s)
#pragma unroll
    for (int n = 0; n < 4; ++n)
      acc[s][n] = (v8f){};

  const int ldRow = tid >> 3;        // 0..31
  const int ldCol = (tid & 7) << 2;  // 0,4,...,28

  for (int k = 0; k < DD; k += 32) {
    // ---- stage 32x32 fp32 A panel -> hi/lo bf16 in LDS (coalesced) ----
    float4 v = *(const float4*)(A + (mBase + ldRow) * DD + k + ldCol);
    float vv[4] = {v.x, v.y, v.z, v.w};
    unsigned short h[4], l[4];
#pragma unroll
    for (int i = 0; i < 4; ++i) {
      h[i] = f2bf_rne(vv[i]);
      l[i] = f2bf_rne(vv[i] - bf2f(h[i]));
    }
    uint2 ph, pl;
    ph.x = (unsigned)h[0] | ((unsigned)h[1] << 16);
    ph.y = (unsigned)h[2] | ((unsigned)h[3] << 16);
    pl.x = (unsigned)l[0] | ((unsigned)l[1] << 16);
    pl.y = (unsigned)l[2] | ((unsigned)l[3] << 16);
    *(uint2*)&Ah[ldRow][ldCol] = ph;
    *(uint2*)&Al[ldRow][ldCol] = pl;
    __syncthreads();

    // ---- A fragments (ISA 16-bit A 16x32 layout) ----
    // lane<16: row=l16,   K pieces {0..7, 16..23}
    // lane>=16: row=l16,  K pieces {8..15, 24..31}
    v16bf afh[2], afl[2];
#pragma unroll
    for (int s = 0; s < 2; ++s) {
      int row = s * 16 + l16;
      int p0 = hi * 8;
      uint4 a0 = *(const uint4*)&Ah[row][p0];
      uint4 a1 = *(const uint4*)&Ah[row][p0 + 16];
      afh[s] = mkfrag(a0, a1);
      uint4 b0 = *(const uint4*)&Al[row][p0];
      uint4 b1 = *(const uint4*)&Al[row][p0 + 16];
      afl[s] = mkfrag(b0, b1);
    }

    // ---- B fragments straight from global: K pieces {8h..8h+7, 16+8h..16+8h+7} of W's row ----
#pragma unroll
    for (int n = 0; n < 4; ++n) {
      int col = wv * 64 + n * 16 + l16;
      const uint4* qh = (const uint4*)(Wh + (size_t)col * DD + k + hi * 8);
      const uint4* ql = (const uint4*)(Wl + (size_t)col * DD + k + hi * 8);
      v16bf bh = mkfrag(qh[0], qh[2]);
      v16bf bl = mkfrag(ql[0], ql[2]);
#pragma unroll
      for (int s = 0; s < 2; ++s) {
        acc[s][n] = __builtin_amdgcn_wmma_f32_16x16x32_bf16(
            false, afh[s], false, bh, (short)0, acc[s][n], false, false);
        acc[s][n] = __builtin_amdgcn_wmma_f32_16x16x32_bf16(
            false, afh[s], false, bl, (short)0, acc[s][n], false, false);
        acc[s][n] = __builtin_amdgcn_wmma_f32_16x16x32_bf16(
            false, afl[s], false, bh, (short)0, acc[s][n], false, false);
      }
    }
    __syncthreads();
  }

  // ---- store: C layout: VGPR i -> row (hi?8:0)+i, col l16 ----
#pragma unroll
  for (int s = 0; s < 2; ++s) {
#pragma unroll
    for (int n = 0; n < 4; ++n) {
      int col = wv * 64 + n * 16 + l16;
#pragma unroll
      for (int i = 0; i < 8; ++i) {
        long row = mBase + s * 16 + hi * 8 + i;
        size_t idx = (size_t)row * DD + col;
        float base = accumulate ? out[idx] : bias[col];
        out[idx] = base + acc[s][n][i];
      }
    }
  }
}

// ---------------------------------------------------------------------------
extern "C" void kernel_launch(void* const* d_in, const int* in_sizes, int n_in,
                              void* d_out, int out_size, void* d_ws,
                              size_t ws_size, hipStream_t stream) {
  (void)in_sizes; (void)n_in; (void)out_size; (void)ws_size;
  const float* x     = (const float*)d_in[0];
  const float* Wself = (const float*)d_in[1];
  const float* Wrel  = (const float*)d_in[2];
  const float* bias  = (const float*)d_in[3];
  const int*   ei    = (const int*)d_in[4];   // [2, E]
  const int*   et    = (const int*)d_in[5];   // [E]
  float* out = (float*)d_out;

  char* ws = (char*)d_ws;
  size_t off = 0;
  auto carve = [&](size_t bytes) {
    char* p = ws + off;
    off = (off + bytes + 255) & ~(size_t)255;
    return p;
  };
  int*            cnt = (int*)carve((size_t)NR * NN * 4);
  float*          inv = (float*)carve((size_t)NR * NN * 4);
  unsigned short* Wh  = (unsigned short*)carve((size_t)(NR + 1) * DD * DD * 2);
  unsigned short* Wl  = (unsigned short*)carve((size_t)(NR + 1) * DD * DD * 2);
  float*          agg = (float*)carve((size_t)NN * DD * 4);

  // weight bf16 hi/lo split
  const size_t wtot = (size_t)(NR + 1) * DD * DD;
  convw_kernel<<<(int)((wtot + 255) / 256), 256, 0, stream>>>(Wself, Wrel, Wh, Wl);

  // per-(relation,dst) degree and 1/max(cnt,1)
  hipMemsetAsync(cnt, 0, (size_t)NR * NN * 4, stream);
  count_kernel<<<(NE + 255) / 256, 256, 0, stream>>>(ei, et, cnt);
  inv_kernel<<<(NR * NN + 255) / 256, 256, 0, stream>>>(cnt, inv);

  // out = x @ W_self^T + bias
  rgcn_gemm<<<NN / 32, 256, 0, stream>>>(x, Wh, Wl, out, bias, 0);

  // out += (mean_r aggregate of x) @ W_r^T, relation by relation
  for (int r = 0; r < NR; ++r) {
    hipMemsetAsync(agg, 0, (size_t)NN * DD * 4, stream);
    scatter_kernel<<<NE / 8, 256, 0, stream>>>(x, ei, et, inv, agg, r);
    rgcn_gemm<<<NN / 32, 256, 0, stream>>>(
        agg, Wh + (size_t)(r + 1) * DD * DD, Wl + (size_t)(r + 1) * DD * DD,
        out, bias, 1);
  }
}
